// GraphSAGE_51891794870978
// MI455X (gfx1250) — matrix operand order repaired and checked by reading, not compiled, not run on hardware
//
#include <hip/hip_runtime.h>
#include <hip/hip_bf16.h>
#include <stdint.h>

#if __has_include(<hip/amd_detail/amd_gfx1250_TDM.h>)
#define TDM_6ARG 1
#endif

typedef __attribute__((ext_vector_type(16))) _Float16 v16h;
typedef __attribute__((ext_vector_type(8)))  _Float16 v8h;
typedef __attribute__((ext_vector_type(8)))  float    v8f;
typedef __attribute__((ext_vector_type(4)))  unsigned int v4u;
typedef __attribute__((ext_vector_type(8)))  int      v8i;
typedef __attribute__((ext_vector_type(4)))  int      v4i;

#define D_HID   128
#define KDIM    256
#define BN_EPS  1e-5f

// ---------------------------------------------------------------- utilities
__global__ void zero_f32(float* __restrict__ p, long long n) {
    long long i = (long long)blockIdx.x * blockDim.x + threadIdx.x;
    long long stride = (long long)gridDim.x * blockDim.x;
    for (; i < n; i += stride) p[i] = 0.0f;
}

// ------------------------------------------------------------- degree / mean
__global__ void degree_kernel(const int* __restrict__ dst, int E, float* __restrict__ deg) {
    int e = blockIdx.x * blockDim.x + threadIdx.x;
    if (e < E) unsafeAtomicAdd(&deg[dst[e]], 1.0f);
}

__global__ void invert_deg_kernel(float* __restrict__ deg, int n) {
    int i = blockIdx.x * blockDim.x + threadIdx.x;
    if (i < n) { float d = deg[i]; deg[i] = (d > 0.0f) ? (1.0f / d) : 0.0f; }
}

// ----------------------------------------------------- edge gather + scatter
// one thread per (edge, 4-float chunk): 32 threads cover one 128-float row
__global__ void scatter_kernel(const int* __restrict__ src, const int* __restrict__ dst,
                               const float* __restrict__ x, float* __restrict__ agg,
                               long long total) {
    long long gid = (long long)blockIdx.x * blockDim.x + threadIdx.x;
    if (gid >= total) return;
    int e = (int)(gid >> 5);
    int g = (int)(gid & 31);
    int s = src[e];
    int d = dst[e];
    const float4 v = *(const float4*)(x + (long long)s * D_HID + g * 4);
    float* a = agg + (long long)d * D_HID + g * 4;
    unsafeAtomicAdd(a + 0, v.x);
    unsafeAtomicAdd(a + 1, v.y);
    unsafeAtomicAdd(a + 2, v.z);
    unsafeAtomicAdd(a + 3, v.w);
}

// ------------------------------------------- pack activations [agg*inv | x] -> f16
__global__ void pack_act_kernel(const float* __restrict__ agg, const float* __restrict__ x,
                                const float* __restrict__ invdeg, _Float16* __restrict__ act,
                                long long n128) {
    long long idx = (long long)blockIdx.x * blockDim.x + threadIdx.x;
    if (idx >= n128) return;
    long long i = idx >> 7;
    int c = (int)(idx & 127);
    float s = invdeg[i];
    act[i * KDIM + c]         = (_Float16)(agg[idx] * s);
    act[i * KDIM + D_HID + c] = (_Float16)(x[idx]);
}

// ------------------------------------- pack weights [Wl;Wr] column-major f16
__global__ void pack_w_kernel(const float* __restrict__ Wl, const float* __restrict__ Wr,
                              _Float16* __restrict__ w16t) {
    int idx = blockIdx.x * blockDim.x + threadIdx.x;     // 256*128
    if (idx >= KDIM * D_HID) return;
    int n = idx & 127;
    int k = idx >> 7;
    float v = (k < D_HID) ? Wl[k * D_HID + n] : Wr[(k - D_HID) * D_HID + n];
    w16t[(long long)n * KDIM + k] = (_Float16)v;
}

// -------------------------------------------------- WMMA GEMM: h = act @ Wt + b
// block = 256 threads = 8 waves; block owns 16 rows, wave w owns cols [16w,16w+16)
// A tile (16x256 f16 = 8KB) staged into LDS once per block by the Tensor Data
// Mover; all 8 waves read it back via ds_load. B (weights) streams from L2.
__global__ __launch_bounds__(256) void gemm_wmma_kernel(const _Float16* __restrict__ act,
                                                        const _Float16* __restrict__ w16t,
                                                        const float* __restrict__ bias,
                                                        float* __restrict__ h, int nrows) {
    __shared__ _Float16 smemA[16 * KDIM];

    int lane  = threadIdx.x & 31;
    int wave  = threadIdx.x >> 5;            // 0..7 -> column tile
    int rowbase = blockIdx.x * 16;
    int rrow  = lane & 15;                   // A row / B col / D col within tile
    int khalf = lane >> 4;                   // 0 or 1 : K half-group select
    int n     = wave * 16 + rrow;            // global output column

#if __has_builtin(__builtin_amdgcn_tensor_load_to_lds)
    if (wave == 0) {
        // --- TDM descriptor: 2D tensor [nrows,256] f16, tile [16,256] at rowbase
        unsigned long long ga =
            (unsigned long long)(uintptr_t)(act + (size_t)rowbase * KDIM);
        unsigned int ldsoff = (unsigned int)(uintptr_t)(void*)smemA;
        v4u g0;
        g0[0] = 1u;                                   // count=1, user descriptor
        g0[1] = ldsoff;                               // lds_addr (bytes)
        g0[2] = (unsigned int)(ga & 0xFFFFFFFFu);     // global_addr[31:0]
        g0[3] = (unsigned int)((ga >> 32) & 0x01FFFFFFu) | 0x80000000u; // [56:32]|type=2
        v8i g1;
        g1[0] = 0x00010000;                           // data_size=1 (2 bytes), no mask
        g1[1] = (int)((KDIM & 0xFFFF) << 16);         // tensor_dim0[15:0]
        g1[2] = (int)(((unsigned)KDIM >> 16) | (((unsigned)nrows & 0xFFFFu) << 16));
        g1[3] = (int)((((unsigned)nrows >> 16) & 0xFFFFu) | ((unsigned)KDIM << 16)); // tile_dim0
        g1[4] = 16;                                   // tile_dim1=16, tile_dim2=0
        g1[5] = KDIM;                                 // tensor_dim0_stride[31:0]
        g1[6] = 0;                                    // stride0 hi, dim1_stride lo
        g1[7] = 0;
        v4i gz = {0, 0, 0, 0};
#ifdef TDM_6ARG
        v8i gz8 = {0, 0, 0, 0, 0, 0, 0, 0};
        __builtin_amdgcn_tensor_load_to_lds(g0, g1, gz, gz, gz8, 0);
#else
        __builtin_amdgcn_tensor_load_to_lds(g0, g1, gz, gz, 0);
#endif
        __builtin_amdgcn_s_wait_tensorcnt(0);
    }
#else
    // fallback: cooperative global->LDS copy
    {
        const v8h* gsrc = (const v8h*)(act + (size_t)rowbase * KDIM);
        v8h* ldst = (v8h*)smemA;
        for (int t = threadIdx.x; t < 16 * KDIM / 8; t += blockDim.x) ldst[t] = gsrc[t];
    }
#endif
    __syncthreads();

    const _Float16* aRow = smemA + rrow * KDIM;
    const _Float16* bcol = w16t + (long long)n * KDIM;

    v8f c = {};
    #pragma unroll
    for (int k = 0; k < KDIM; k += 32) {
        // A 16x32 f16 layout: lanes 0-15 K={0..7,16..23}, lanes 16-31 K={8..15,24..31}
        v8h alo = *(const v8h*)(aRow + k + khalf * 8);
        v8h ahi = *(const v8h*)(aRow + k + 16 + khalf * 8);
        v16h a = __builtin_shufflevector(alo, ahi,
                                         0, 1, 2, 3, 4, 5, 6, 7,
                                         8, 9, 10, 11, 12, 13, 14, 15);
        // B 32x16 f16: lanes 0-15 K={0..7,16..23} of column n, lanes 16-31 K={8..15,24..31}
        v8h blo = *(const v8h*)(bcol + k + khalf * 8);
        v8h bhi = *(const v8h*)(bcol + k + 16 + khalf * 8);
        v16h b = __builtin_shufflevector(blo, bhi,
                                         0, 1, 2, 3, 4, 5, 6, 7,
                                         8, 9, 10, 11, 12, 13, 14, 15);
        c = __builtin_amdgcn_wmma_f32_16x16x32_f16(
                /*neg_a=*/false, a, /*neg_b=*/false, b,
                /*c_mod=*/(short)0, c, /*reuse_a=*/false, /*reuse_b=*/false);
    }

    float bv = bias[n];
    #pragma unroll
    for (int r = 0; r < 8; ++r) {            // D: M = r + 8*khalf, N = lane&15
        int m = rowbase + r + (khalf << 3);
        if (m < nrows) h[(long long)m * D_HID + n] = c[r] + bv;
    }
}

// ------------------------------------------------------------- batchnorm
__global__ void bn_stats_kernel(const float* __restrict__ h, float* __restrict__ s1,
                                float* __restrict__ s2, int n) {
    int c = threadIdx.x;                      // 128 threads = 128 columns
    float a = 0.0f, b = 0.0f;
    for (int i = blockIdx.x; i < n; i += gridDim.x) {
        float v = h[(long long)i * D_HID + c];
        a += v; b += v * v;
    }
    unsafeAtomicAdd(&s1[c], a);
    unsafeAtomicAdd(&s2[c], b);
}

__global__ void bn_finalize_kernel(const float* __restrict__ s1, const float* __restrict__ s2,
                                   const float* __restrict__ gamma, const float* __restrict__ beta,
                                   float* __restrict__ scale, float* __restrict__ shift, int n) {
    int c = threadIdx.x;
    float inv_n = 1.0f / (float)n;
    float mu  = s1[c] * inv_n;
    float var = s2[c] * inv_n - mu * mu;
    float r   = rsqrtf(var + BN_EPS);
    float sc  = gamma[c] * r;
    scale[c] = sc;
    shift[c] = beta[c] - mu * sc;
}

__global__ void bn_apply_kernel(const float* __restrict__ h, const float* __restrict__ scale,
                                const float* __restrict__ shift, float* __restrict__ out,
                                long long n128, int relu) {
    long long idx = (long long)blockIdx.x * blockDim.x + threadIdx.x;
    if (idx >= n128) return;
    int c = (int)(idx & 127);
    float v = h[idx] * scale[c] + shift[c];
    if (relu) v = fmaxf(v, 0.0f);
    out[idx] = v;
}

// ------------------------------------------------------------- classifier
__global__ void classifier_kernel(const float* __restrict__ emb, const float* __restrict__ Wc,
                                  const float* __restrict__ bc, float* __restrict__ out, int n) {
    int i = blockIdx.x * blockDim.x + threadIdx.x;
    if (i >= n) return;
    const float* xr = emb + (long long)i * D_HID;
    float s0 = bc[0], s1 = bc[1];
    #pragma unroll 8
    for (int k = 0; k < D_HID; ++k) {
        float v = xr[k];
        s0 += v * Wc[k * 2 + 0];
        s1 += v * Wc[k * 2 + 1];
    }
    out[(long long)i * 2 + 0] = s0;
    out[(long long)i * 2 + 1] = s1;
}

// =========================================================================
extern "C" void kernel_launch(void* const* d_in, const int* in_sizes, int n_in,
                              void* d_out, int out_size, void* d_ws, size_t ws_size,
                              hipStream_t stream) {
    const float* x     = (const float*)d_in[0];
    const int*   ei    = (const int*)d_in[1];
    const float* Wl    = (const float*)d_in[2];
    const float* bl    = (const float*)d_in[3];
    const float* Wr    = (const float*)d_in[4];
    const float* gamma = (const float*)d_in[5];
    const float* beta  = (const float*)d_in[6];
    const float* Wc    = (const float*)d_in[7];
    const float* bc    = (const float*)d_in[8];

    const int N = in_sizes[0] / D_HID;       // 50000
    const int E = in_sizes[1] / 2;           // 800000
    const int* src = ei;
    const int* dst = ei + E;

    float* out = (float*)d_out;              // [N,2]
    float* emb = out + (long long)N * 2;     // [N,128]

    // -------- workspace carve-out
    char* base = (char*)d_ws;
    size_t off = 0;
    auto carve = [&](size_t bytes) -> char* {
        char* p = base + off;
        off = (off + bytes + 255) & ~(size_t)255;
        return p;
    };
    float*    invdeg = (float*)   carve((size_t)N * sizeof(float));
    float*    agg    = (float*)   carve((size_t)N * D_HID * sizeof(float));
    float*    hbuf   = (float*)   carve((size_t)N * D_HID * sizeof(float));
    float*    xbuf   = (float*)   carve((size_t)N * D_HID * sizeof(float));
    _Float16* act    = (_Float16*)carve((size_t)N * KDIM * sizeof(_Float16));
    _Float16* w16t   = (_Float16*)carve((size_t)KDIM * D_HID * sizeof(_Float16));
    float*    s1     = (float*)   carve(128 * sizeof(float));
    float*    s2     = (float*)   carve(128 * sizeof(float));
    float*    scale  = (float*)   carve(128 * sizeof(float));
    float*    shift  = (float*)   carve(128 * sizeof(float));
    (void)ws_size; (void)n_in; (void)out_size;

    const long long n128 = (long long)N * D_HID;
    const int mtiles = (N + 15) / 16;

    // -------- mean-aggregation degree (layer invariant)
    zero_f32<<<512, 256, 0, stream>>>(invdeg, N);
    degree_kernel<<<(E + 255) / 256, 256, 0, stream>>>(dst, E, invdeg);
    invert_deg_kernel<<<(N + 255) / 256, 256, 0, stream>>>(invdeg, N);

    for (int layer = 0; layer < 3; ++layer) {
        const float* xcur  = (layer == 0) ? x : xbuf;
        float*       xnext = (layer == 2) ? emb : xbuf;

        zero_f32<<<2048, 256, 0, stream>>>(agg, n128);
        zero_f32<<<1, 256, 0, stream>>>(s1, 256);     // s1+s2 contiguous

        long long sc_total = (long long)E * 32;
        scatter_kernel<<<(int)((sc_total + 255) / 256), 256, 0, stream>>>(
            src, dst, xcur, agg, sc_total);

        pack_act_kernel<<<(int)((n128 + 255) / 256), 256, 0, stream>>>(
            agg, xcur, invdeg, act, n128);
        pack_w_kernel<<<(KDIM * D_HID + 255) / 256, 256, 0, stream>>>(
            Wl + (long long)layer * D_HID * D_HID,
            Wr + (long long)layer * D_HID * D_HID, w16t);

        gemm_wmma_kernel<<<mtiles, 256, 0, stream>>>(
            act, w16t, bl + layer * D_HID, hbuf, N);

        bn_stats_kernel<<<512, 128, 0, stream>>>(hbuf, s1, s2, N);
        bn_finalize_kernel<<<1, 128, 0, stream>>>(
            s1, s2, gamma + layer * D_HID, beta + layer * D_HID, scale, shift, N);
        bn_apply_kernel<<<(int)((n128 + 255) / 256), 256, 0, stream>>>(
            hbuf, scale, shift, xnext, n128, layer < 2 ? 1 : 0);
    }

    classifier_kernel<<<(N + 255) / 256, 256, 0, stream>>>(emb, Wc, bc, out, N);
}
